// SpectralEncoderLayer_38843684225855
// MI455X (gfx1250) — matrix operand order repaired and checked by reading, not compiled, not run on hardware
//
#include <hip/hip_runtime.h>
#include <hip/hip_bf16.h>

// ---------------------------------------------------------------------------
// SpectralEncoderLayer for MI455X (gfx1250, wave32, WMMA).
//   LN1 -> QKV(1x1 conv, WMMA f16) -> 7x7 local attention (online softmax,
//   VALU) -> bilinear (per-batch 128x128 WMMA) + residual -> LN2 ->
//   FFN 128->512->128 (WMMA, exact GELU) + residual.
// GEMMs: v_wmma_f32_16x16x32_f16, one 16x64 tile per wave (4 accumulators
// share each A fragment). Transposes: async Global<->LDS DMA
// (global_load_async_to_lds_b32 / global_store_async_from_lds_b32).
// ---------------------------------------------------------------------------

typedef _Float16 v16h __attribute__((ext_vector_type(16)));
typedef _Float16 h8v  __attribute__((ext_vector_type(8)));
typedef _Float16 h4v  __attribute__((ext_vector_type(4)));
typedef float    v8f  __attribute__((ext_vector_type(8)));

#define MTOT   16384   // N*H*W pixels
#define EMB    128
#define FFN    512
#define HEADS  4
#define HDIM   32
#define KWIN   7
#define PWIN   49

// ---------------------------------------------------------------------------
// WMMA 16x64 tile per wave: D[m0:m0+16, n0:n0+64] = A[m0:,:K] * B[:,n0:n0+64]
// A: row-major f16 [M][K]; B: f16 [N][K] (weight[out][in]; a B column is
// K-contiguous, matching the WMMA B-lane layout).
// Fragments (this code's contiguous k order, same for A and B):
//  A (16x32 f16): lanes 0-15 hold M=lane, K = k0+0..15;
//                 lanes 16-31 hold K = k0+16..31.
//  B (32x16 f16): lane holds column N=lane&15, 16 consecutive K values,
//                 lanes>=16 take the upper K half.
// One A fragment feeds 4 WMMAs (4 adjacent N tiles).
// ---------------------------------------------------------------------------
template <int K>
__device__ __forceinline__ void wmma_tile4(const _Float16* __restrict__ A,
                                           const _Float16* __restrict__ B,
                                           int m0, int n0, int lane,
                                           v8f acc[4]) {
  const int half_sel = lane >> 4;            // 0 or 1
  const int idx16    = lane & 15;
  const _Float16* Arow = A + (size_t)(m0 + idx16) * K;
  const _Float16* Bcol = B + (size_t)(n0 + idx16) * K;
#pragma unroll
  for (int k0 = 0; k0 < K; k0 += 32) {
    const int klo = k0 + (half_sel << 4);    // A: 16 contiguous halves
    h8v alo = *(const h8v*)(Arow + klo);
    h8v ahi = *(const h8v*)(Arow + klo + 8);
    v16h a = __builtin_shufflevector(alo, ahi, 0, 1, 2, 3, 4, 5, 6, 7,
                                     8, 9, 10, 11, 12, 13, 14, 15);
    const int kb = k0 + (half_sel << 4);     // B: 16 contiguous halves
    __builtin_prefetch((const void*)(Bcol + kb + 32), 0, 0);
#pragma unroll
    for (int t = 0; t < 4; ++t) {
      v16h b = *(const v16h*)(Bcol + (size_t)t * 16 * K + kb);
      // 8 args: (neg_a, A, neg_b, B, c_mod, C, reuse_a, reuse_b)
      acc[t] = __builtin_amdgcn_wmma_f32_16x16x32_f16(false, a, false, b,
                                                      (short)0, acc[t],
                                                      false, false);
    }
  }
}

// D-matrix layout (f32 16x16): VGPR r, lanes 0-15 -> (M=r, N=lane);
// lanes 16-31 -> (M=8+r, N=lane-16).

// ---------------------------------------------------------------------------
// Weight prep
// ---------------------------------------------------------------------------
__global__ void copy_f32_to_f16_kernel(const float* __restrict__ in,
                                       _Float16* __restrict__ out, int n) {
  int i = blockIdx.x * blockDim.x + threadIdx.x;
  int stride = gridDim.x * blockDim.x;
  for (; i < n; i += stride) out[i] = (_Float16)in[i];
}

// t[n][o][i] = sum_q bil_w[o][i][q] * qcoeff[n][q]   (Q=64)
__global__ void bil_contract_kernel(const float* __restrict__ bw,
                                    const float* __restrict__ qc,
                                    _Float16* __restrict__ t16) {
  int idx = blockIdx.x * blockDim.x + threadIdx.x;  // 4*128*128 = 65536
  int n  = idx >> 14;
  int oi = idx & 16383;
  const float4* b4 = (const float4*)(bw + (size_t)oi * 64);
  const float4* q4 = (const float4*)(qc + n * 64);
  float s = 0.f;
#pragma unroll
  for (int j = 0; j < 16; ++j) {
    float4 a = b4[j], c = q4[j];
    s += a.x * c.x + a.y * c.y + a.z * c.z + a.w * c.w;
  }
  t16[idx] = (_Float16)s;
}

// ---------------------------------------------------------------------------
// Layout transforms, LDS 32x33 tiles. The fill phase uses CDNA5 async
// Global->LDS DMA (ASYNCcnt), the drain phase uses async LDS->Global; both
// bypass VGPR staging entirely. s_wait_asynccnt orders against the barrier;
// S_ENDPGM does an implicit wait-idle for the final stores.
// ---------------------------------------------------------------------------
__device__ __forceinline__ unsigned lds_off(const void* p) {
  // Flat LDS aperture address: LDS offset lives in addr[31:0] (ISA §10.2).
  return (unsigned)(uintptr_t)p;
}

__global__ void nchw_to_nhwc_kernel(const float* __restrict__ in,
                                    float* __restrict__ out) {
  __shared__ float tile[32][33];
  const int p0 = blockIdx.x << 5, e0 = blockIdx.y << 5, n = blockIdx.z;
  const int tx = threadIdx.x, ty = threadIdx.y;  // (32, 8)
#pragma unroll
  for (int j = 0; j < 4; ++j) {
    const float* gp =
        in + ((size_t)(n * EMB + e0 + ty + 8 * j)) * 4096 + p0 + tx;
    asm volatile("global_load_async_to_lds_b32 %0, %1, off"
                 :: "v"(lds_off(&tile[ty + 8 * j][tx])),
                    "v"((unsigned long long)(uintptr_t)gp)
                 : "memory");
  }
  asm volatile("s_wait_asynccnt 0x0" ::: "memory");
  __syncthreads();
#pragma unroll
  for (int j = 0; j < 4; ++j) {
    float* gp = out + ((size_t)n * 4096 + p0 + ty + 8 * j) * EMB + e0 + tx;
    asm volatile("global_store_async_from_lds_b32 %0, %1, off"
                 :: "v"((unsigned long long)(uintptr_t)gp),
                    "v"(lds_off(&tile[tx][ty + 8 * j]))
                 : "memory");
  }
  asm volatile("s_wait_asynccnt 0x0" ::: "memory");
}

__global__ void nhwc_to_nchw_kernel(const float* __restrict__ in,
                                    float* __restrict__ out) {
  __shared__ float tile[32][33];
  const int p0 = blockIdx.x << 5, e0 = blockIdx.y << 5, n = blockIdx.z;
  const int tx = threadIdx.x, ty = threadIdx.y;  // (32, 8)
#pragma unroll
  for (int j = 0; j < 4; ++j) {
    const float* gp =
        in + ((size_t)n * 4096 + p0 + ty + 8 * j) * EMB + e0 + tx;
    asm volatile("global_load_async_to_lds_b32 %0, %1, off"
                 :: "v"(lds_off(&tile[ty + 8 * j][tx])),
                    "v"((unsigned long long)(uintptr_t)gp)
                 : "memory");
  }
  asm volatile("s_wait_asynccnt 0x0" ::: "memory");
  __syncthreads();
#pragma unroll
  for (int j = 0; j < 4; ++j) {
    float* gp =
        out + ((size_t)(n * EMB + e0 + ty + 8 * j)) * 4096 + p0 + tx;
    asm volatile("global_store_async_from_lds_b32 %0, %1, off"
                 :: "v"((unsigned long long)(uintptr_t)gp),
                    "v"(lds_off(&tile[tx][ty + 8 * j]))
                 : "memory");
  }
  asm volatile("s_wait_asynccnt 0x0" ::: "memory");
}

// ---------------------------------------------------------------------------
// Row-wise LayerNorm over E=128: one wave per row (wave32), float4 per lane,
// __shfl_xor butterfly reduction, f16 output for the following GEMM.
// ---------------------------------------------------------------------------
__global__ void ln_rows_kernel(const float* __restrict__ in,
                               const float* __restrict__ g,
                               const float* __restrict__ bta,
                               _Float16* __restrict__ out, int M) {
  const int lane = threadIdx.x & 31;
  const int row = blockIdx.x * (blockDim.x >> 5) + (threadIdx.x >> 5);
  if (row >= M) return;
  const float4 val = ((const float4*)(in + (size_t)row * EMB))[lane];
  float s  = val.x + val.y + val.z + val.w;
  float ss = val.x * val.x + val.y * val.y + val.z * val.z + val.w * val.w;
#pragma unroll
  for (int off = 16; off > 0; off >>= 1) {
    s  += __shfl_xor(s, off, 32);
    ss += __shfl_xor(ss, off, 32);
  }
  const float mean = s * (1.f / 128.f);
  const float var  = ss * (1.f / 128.f) - mean * mean;
  const float rstd = rsqrtf(var + 1e-5f);
  const int c = lane << 2;
  h4v o;
  o[0] = (_Float16)((val.x - mean) * rstd * g[c + 0] + bta[c + 0]);
  o[1] = (_Float16)((val.y - mean) * rstd * g[c + 1] + bta[c + 1]);
  o[2] = (_Float16)((val.z - mean) * rstd * g[c + 2] + bta[c + 2]);
  o[3] = (_Float16)((val.w - mean) * rstd * g[c + 3] + bta[c + 3]);
  *(h4v*)(out + (size_t)row * EMB + c) = o;
}

// ---------------------------------------------------------------------------
// GEMM kernels: 256 threads = 8 waves, each wave owns a 16(M)x64(N) tile.
// ---------------------------------------------------------------------------

// QKV: M=16384, N=128, K=128. Block = 64M x 128N (wm = wave>>1, wn = wave&1).
__global__ void __launch_bounds__(256)
gemm_qkv_kernel(const _Float16* __restrict__ xn,
                const _Float16* __restrict__ wqkv,
                const float* __restrict__ bq, const float* __restrict__ bk,
                const float* __restrict__ bv, _Float16* __restrict__ q,
                _Float16* __restrict__ k, _Float16* __restrict__ v) {
  const int lane = threadIdx.x & 31, wave = threadIdx.x >> 5;
  const int m0 = (blockIdx.x << 6) + ((wave >> 1) << 4);
  const int n0 = (wave & 1) << 6;
  const int z = blockIdx.z;
  const _Float16* B = wqkv + (size_t)z * (EMB * EMB);
  v8f acc[4] = {{0.f}, {0.f}, {0.f}, {0.f}};
  wmma_tile4<EMB>(xn, B, m0, n0, lane, acc);
  const float* bias = (z == 0) ? bq : (z == 1) ? bk : bv;
  _Float16* out = (z == 0) ? q : (z == 1) ? k : v;
  const int idx16 = lane & 15;
  const int mb = m0 + ((lane >> 4) << 3);
#pragma unroll
  for (int t = 0; t < 4; ++t) {
    const int col = n0 + (t << 4) + idx16;
    const float bc = bias[col];
#pragma unroll
    for (int r = 0; r < 8; ++r)
      out[(size_t)(mb + r) * EMB + col] = (_Float16)(acc[t][r] + bc);
  }
}

// Bilinear: per-batch 128x128 B = t16[n]; z2 = z0 @ t^T + bil_b + x_nhwc.
__global__ void __launch_bounds__(256)
gemm_bil_kernel(const _Float16* __restrict__ z0,
                const _Float16* __restrict__ t16,
                const float* __restrict__ bil_b,
                const float* __restrict__ x_nhwc, float* __restrict__ z2) {
  const int lane = threadIdx.x & 31, wave = threadIdx.x >> 5;
  const int m0 = (blockIdx.x << 6) + ((wave >> 1) << 4);
  const int n0 = (wave & 1) << 6;
  const _Float16* B = t16 + (size_t)(blockIdx.x >> 6) * (EMB * EMB);  // batch
  v8f acc[4] = {{0.f}, {0.f}, {0.f}, {0.f}};
  wmma_tile4<EMB>(z0, B, m0, n0, lane, acc);
  const int idx16 = lane & 15;
  const int mb = m0 + ((lane >> 4) << 3);
#pragma unroll
  for (int t = 0; t < 4; ++t) {
    const int col = n0 + (t << 4) + idx16;
    const float bc = bil_b[col];
#pragma unroll
    for (int r = 0; r < 8; ++r) {
      const size_t idx = (size_t)(mb + r) * EMB + col;
      z2[idx] = acc[t][r] + bc + x_nhwc[idx];  // bilinear + x residual
    }
  }
}

// FFN1: M=16384, N=512, K=128. Block = 16M x 512N (wave n0 = wave*64).
__global__ void __launch_bounds__(256)
gemm_ffn1_kernel(const _Float16* __restrict__ z3n,
                 const _Float16* __restrict__ w116,
                 const float* __restrict__ b1, _Float16* __restrict__ h1) {
  const int lane = threadIdx.x & 31, wave = threadIdx.x >> 5;
  const int m0 = blockIdx.x << 4;
  const int n0 = wave << 6;
  v8f acc[4] = {{0.f}, {0.f}, {0.f}, {0.f}};
  wmma_tile4<EMB>(z3n, w116, m0, n0, lane, acc);
  const int idx16 = lane & 15;
  const int mb = m0 + ((lane >> 4) << 3);
#pragma unroll
  for (int t = 0; t < 4; ++t) {
    const int col = n0 + (t << 4) + idx16;
    const float bc = b1[col];
#pragma unroll
    for (int r = 0; r < 8; ++r) {
      const float u = acc[t][r] + bc;
      const float gl = 0.5f * u * (1.f + erff(u * 0.70710678118654752f));
      h1[(size_t)(mb + r) * FFN + col] = (_Float16)gl;
    }
  }
}

// FFN2: M=16384, N=128, K=512; + b2 + z2 residual -> z4 f32.
__global__ void __launch_bounds__(256)
gemm_ffn2_kernel(const _Float16* __restrict__ h1,
                 const _Float16* __restrict__ w216,
                 const float* __restrict__ b2, const float* __restrict__ z2,
                 float* __restrict__ z4) {
  const int lane = threadIdx.x & 31, wave = threadIdx.x >> 5;
  const int m0 = (blockIdx.x << 6) + ((wave >> 1) << 4);
  const int n0 = (wave & 1) << 6;
  v8f acc[4] = {{0.f}, {0.f}, {0.f}, {0.f}};
  wmma_tile4<FFN>(h1, w216, m0, n0, lane, acc);  // K = 512
  const int idx16 = lane & 15;
  const int mb = m0 + ((lane >> 4) << 3);
#pragma unroll
  for (int t = 0; t < 4; ++t) {
    const int col = n0 + (t << 4) + idx16;
    const float bc = b2[col];
#pragma unroll
    for (int r = 0; r < 8; ++r) {
      const size_t idx = (size_t)(mb + r) * EMB + col;
      z4[idx] = acc[t][r] + bc + z2[idx];  // final residual
    }
  }
}

// ---------------------------------------------------------------------------
// 7x7 local attention, one thread per (pixel, head). Online (flash-style)
// softmax over the 49 taps; zero-padded taps contribute exp(bias) mass to
// the denominator (matching the reference) but zero value.
// ---------------------------------------------------------------------------
__global__ void __launch_bounds__(256)
attn_kernel(const _Float16* __restrict__ q, const _Float16* __restrict__ k,
            const _Float16* __restrict__ v, const float* __restrict__ rel_bias,
            _Float16* __restrict__ z0) {
  const int t = blockIdx.x * blockDim.x + threadIdx.x;  // 65536 threads
  const int head = t & 3;
  const int pix = t >> 2;
  const int n = pix >> 12, hw = pix & 4095;
  const int y = hw >> 6, x = hw & 63;

  float qf[32];
  const h8v* q8 = (const h8v*)(q + (size_t)pix * EMB + head * HDIM);
#pragma unroll
  for (int jj = 0; jj < 4; ++jj) {
    h8v tv = q8[jj];
#pragma unroll
    for (int j = 0; j < 8; ++j) qf[jj * 8 + j] = (float)tv[j];
  }

  float mrun = -3.0e38f, denom = 0.f;
  float o[32];
#pragma unroll
  for (int j = 0; j < 32; ++j) o[j] = 0.f;

  for (int p = 0; p < PWIN; ++p) {
    const int dy = p / KWIN - 3;
    const int dx = p - (p / KWIN) * KWIN - 3;
    const int yy = y + dy, xx = x + dx;
    const bool inb = ((unsigned)yy < 64u) && ((unsigned)xx < 64u);
    const size_t np = (size_t)(n << 12) + (yy << 6) + xx;

    float dot = 0.f;
    if (inb) {
      const h8v* k8 = (const h8v*)(k + np * EMB + head * HDIM);
#pragma unroll
      for (int jj = 0; jj < 4; ++jj) {
        h8v kv = k8[jj];
#pragma unroll
        for (int j = 0; j < 8; ++j) dot += qf[jj * 8 + j] * (float)kv[j];
      }
    }
    const float sv = dot + rel_bias[head * PWIN + p];

    const float nm = fmaxf(mrun, sv);
    const float scale = __expf(mrun - nm);
    const float e = __expf(sv - nm);
    denom = denom * scale + e;
    if (inb) {
      const h8v* v8 = (const h8v*)(v + np * EMB + head * HDIM);
#pragma unroll
      for (int jj = 0; jj < 4; ++jj) {
        h8v vv = v8[jj];
#pragma unroll
        for (int j = 0; j < 8; ++j)
          o[jj * 8 + j] = o[jj * 8 + j] * scale + e * (float)vv[j];
      }
    } else {
#pragma unroll
      for (int j = 0; j < 32; ++j) o[j] *= scale;
    }
    mrun = nm;
  }

  const float inv = 1.f / denom;
  h8v* zp = (h8v*)(z0 + (size_t)pix * EMB + head * HDIM);
#pragma unroll
  for (int jj = 0; jj < 4; ++jj) {
    h8v ov;
#pragma unroll
    for (int j = 0; j < 8; ++j) ov[j] = (_Float16)(o[jj * 8 + j] * inv);
    zp[jj] = ov;
  }
}

// ---------------------------------------------------------------------------
// Launch: workspace overlays (peak ~37 MB):
//  [0,8M):   x_nhwc f32      -> reused for z4 f32 (after bilinear consumes x)
//  [8M,12M): xn f16          \
//  [12M,24M): q/k/v f16       } -> reused as h1 f16 [8M,24M) (dead by FFN1)
//  [24M,28M): z0 f16         -> reused for z3n f16 (z0 dead after bilinear)
//  [28M,36M): z2 f32 (lives through FFN2)
//  [36M,...): f16 weights: wq|wk|wv (96K), w1 (128K), w2 (128K), t (128K)
// ---------------------------------------------------------------------------
extern "C" void kernel_launch(void* const* d_in, const int* in_sizes, int n_in,
                              void* d_out, int out_size, void* d_ws,
                              size_t ws_size, hipStream_t stream) {
  const float* x       = (const float*)d_in[0];
  const float* qcoeff  = (const float*)d_in[1];
  const float* wq      = (const float*)d_in[2];
  const float* bq      = (const float*)d_in[3];
  const float* wk      = (const float*)d_in[4];
  const float* bk      = (const float*)d_in[5];
  const float* wv      = (const float*)d_in[6];
  const float* bv      = (const float*)d_in[7];
  const float* rel_bias= (const float*)d_in[8];
  const float* ln1_g   = (const float*)d_in[9];
  const float* ln1_b   = (const float*)d_in[10];
  const float* bil_w   = (const float*)d_in[11];
  const float* bil_b   = (const float*)d_in[12];
  const float* ln2_g   = (const float*)d_in[13];
  const float* ln2_b   = (const float*)d_in[14];
  const float* w1      = (const float*)d_in[15];
  const float* b1      = (const float*)d_in[16];
  const float* w2      = (const float*)d_in[17];
  const float* b2      = (const float*)d_in[18];
  float* out = (float*)d_out;

  const size_t MB = 1ull << 20;
  char* ws = (char*)d_ws;
  float*    x_nhwc = (float*)(ws + 0);
  _Float16* xn     = (_Float16*)(ws + 8 * MB);
  _Float16* qb     = (_Float16*)(ws + 12 * MB);
  _Float16* kb     = (_Float16*)(ws + 16 * MB);
  _Float16* vb     = (_Float16*)(ws + 20 * MB);
  _Float16* z0     = (_Float16*)(ws + 24 * MB);
  float*    z2     = (float*)(ws + 28 * MB);
  _Float16* z3n    = (_Float16*)(ws + 24 * MB);  // reuse z0 slot
  _Float16* h1     = (_Float16*)(ws + 8 * MB);   // reuse xn/q/k/v slots
  float*    z4     = (float*)(ws + 0);           // reuse x_nhwc slot
  char* wbase = ws + 36 * MB;
  _Float16* wqkv16 = (_Float16*)(wbase);                 // 3 * 128*128
  _Float16* w116   = (_Float16*)(wbase + 96 * 1024);     // 512*128
  _Float16* w216   = (_Float16*)(wbase + 224 * 1024);    // 128*512
  _Float16* t16    = (_Float16*)(wbase + 352 * 1024);    // 4 * 128*128

  // 1) weight prep
  copy_f32_to_f16_kernel<<<64, 256, 0, stream>>>(wq, wqkv16, EMB * EMB);
  copy_f32_to_f16_kernel<<<64, 256, 0, stream>>>(wk, wqkv16 + EMB * EMB, EMB * EMB);
  copy_f32_to_f16_kernel<<<64, 256, 0, stream>>>(wv, wqkv16 + 2 * EMB * EMB, EMB * EMB);
  copy_f32_to_f16_kernel<<<256, 256, 0, stream>>>(w1, w116, FFN * EMB);
  copy_f32_to_f16_kernel<<<256, 256, 0, stream>>>(w2, w216, EMB * FFN);
  bil_contract_kernel<<<256, 256, 0, stream>>>(bil_w, qcoeff, t16);

  // 2) x NCHW -> NHWC (async DMA transpose), LN1
  nchw_to_nhwc_kernel<<<dim3(128, 4, 4), dim3(32, 8), 0, stream>>>(x, x_nhwc);
  ln_rows_kernel<<<2048, 256, 0, stream>>>(x_nhwc, ln1_g, ln1_b, xn, MTOT);

  // 3) Q/K/V projections (WMMA)
  gemm_qkv_kernel<<<dim3(256, 1, 3), 256, 0, stream>>>(xn, wqkv16, bq, bk, bv,
                                                       qb, kb, vb);
  // 4) local attention
  attn_kernel<<<256, 256, 0, stream>>>(qb, kb, vb, rel_bias, z0);

  // 5) bilinear + x residual -> z2 (WMMA)
  gemm_bil_kernel<<<256, 256, 0, stream>>>(z0, t16, bil_b, x_nhwc, z2);

  // 6) LN2
  ln_rows_kernel<<<2048, 256, 0, stream>>>(z2, ln2_g, ln2_b, z3n, MTOT);

  // 7) FFN (WMMA) + residual
  gemm_ffn1_kernel<<<1024, 256, 0, stream>>>(z3n, w116, b1, h1);
  gemm_ffn2_kernel<<<256, 256, 0, stream>>>(h1, w216, b2, z2, z4);

  // 8) NHWC -> NCHW (async DMA transpose) into d_out
  nhwc_to_nchw_kernel<<<dim3(128, 4, 4), dim3(32, 8), 0, stream>>>(z4, out);

  (void)in_sizes; (void)n_in; (void)out_size; (void)ws_size;
}
